// scdEGA_45741401702495
// MI455X (gfx1250) — compile-verified, matrix operand order checked by reading, not run on hardware
//
#include <hip/hip_runtime.h>
#include <hip/hip_bf16.h>

// ---------------------------------------------------------------------------
// scGAE-style graph-attention autoencoder for MI455X (gfx1250, wave32, WMMA).
// All GEMMs run on v_wmma_f32_16x16x32_bf16 (bf16 operands, f32 accumulate).
// ---------------------------------------------------------------------------

#define NC 3000   // cells
#define GC 1000   // genes
#define MC 4000   // decoder nodes (N + G)

typedef __attribute__((ext_vector_type(16))) __bf16 v16bf;
typedef __attribute__((ext_vector_type(8)))  float  v8f;

__device__ __forceinline__ unsigned short f2bf(float f) {
  unsigned int u = __float_as_uint(f);
  unsigned int r = (u + 0x7FFFu + ((u >> 16) & 1u)) >> 16;   // RNE
  return (unsigned short)r;
}
__device__ __forceinline__ unsigned int pack2bf(float a, float b) {
  return (unsigned int)f2bf(a) | ((unsigned int)f2bf(b) << 16);
}

// ---------------------------------------------------------------------------
// WMMA GEMM:  C[M,N], A is MxK row-major (lda),
//   transB==0 : B is KxN row-major (ldb)
//   transB==1 : B is NxK row-major (ldb)   (i.e. C = A * B^T)
// Optional bias (per column) + ReLU epilogue.
// Block: 128x128 tile, 256 threads = 8 waves (2x4), wave tile 64x32.
// ---------------------------------------------------------------------------
#define BM 128
#define BN 128
#define BK 32
#define LDSK (BK + 8)

__device__ __forceinline__ v16bf load_frag(const unsigned short* rowp, int half) {
  // ISA 7.12.2 16-bit A/B fragment: VGPR v<4 -> K = 2v + 8*half,
  // v>=4 -> K = 16 + 2(v-4) + 8*half ; each VGPR packs (K, K+1).
  v16bf f;
#pragma unroll
  for (int v = 0; v < 8; ++v) {
    int k0 = ((v < 4) ? (v * 2) : (16 + (v - 4) * 2)) + half * 8;
    f[2 * v]     = __builtin_bit_cast(__bf16, rowp[k0]);
    f[2 * v + 1] = __builtin_bit_cast(__bf16, rowp[k0 + 1]);
  }
  return f;
}

// One 128x128x32 MMA step from LDS (8 WMMAs per wave).
__device__ __forceinline__ void mma_step(const unsigned short As[BM][LDSK],
                                         const unsigned short Bs[BN][LDSK],
                                         v8f acc[4][2],
                                         int waveM, int waveN, int l16, int half) {
  v16bf af[4], bf[2];
#pragma unroll
  for (int tm = 0; tm < 4; ++tm)
    af[tm] = load_frag(&As[waveM * 64 + tm * 16 + l16][0], half);
#pragma unroll
  for (int tn = 0; tn < 2; ++tn)
    bf[tn] = load_frag(&Bs[waveN * 32 + tn * 16 + l16][0], half);
#pragma unroll
  for (int tm = 0; tm < 4; ++tm)
#pragma unroll
    for (int tn = 0; tn < 2; ++tn)
      acc[tm][tn] = __builtin_amdgcn_wmma_f32_16x16x32_bf16(
          false, af[tm], false, bf[tn], (short)0, acc[tm][tn], false, false);
}

__global__ __launch_bounds__(256) void gemm_bf16_wmma(
    const float* __restrict__ A, const float* __restrict__ B,
    float* __restrict__ C, const float* __restrict__ bias,
    int M, int N, int K, int lda, int ldb, int ldc, int transB, int doRelu) {
  __shared__ unsigned short As[BM][LDSK];   // [m][k]
  __shared__ unsigned short Bs[BN][LDSK];   // [n][k]

  const int tid   = threadIdx.x;
  const int wave  = tid >> 5;
  const int lane  = tid & 31;
  const int waveM = wave >> 2;   // 0..1  -> 64-row strip
  const int waveN = wave & 3;    // 0..3  -> 32-col strip
  const int half  = lane >> 4;
  const int l16   = lane & 15;

  const int blockM = blockIdx.y * BM;
  const int blockN = blockIdx.x * BN;
  const bool interior = (blockM + BM <= M) && (blockN + BN <= N);

  v8f acc[4][2];
#pragma unroll
  for (int a = 0; a < 4; ++a)
#pragma unroll
    for (int b = 0; b < 2; ++b)
#pragma unroll
      for (int e = 0; e < 8; ++e) acc[a][b][e] = 0.0f;

  const int kfull  = K / BK;               // full (unguarded) k-steps
  const int ksteps = (K + BK - 1) / BK;

  // Branch-free guarded tile load (cold path: edge blocks and K-tail).
  // Addresses are clamped into range; OOB values are zero-selected.
  auto load_guarded = [&](int k0) {
#pragma unroll 1
    for (int i = 0; i < (BM * BK) / 256; ++i) {
      int idx = tid + i * 256;
      int m = idx >> 5, k = idx & 31;
      int gm = blockM + m, gk = k0 + k;
      int cm = gm < M ? gm : M - 1;
      int ck = gk < K ? gk : K - 1;
      float v = A[(size_t)cm * lda + ck];
      if (gm >= M || gk >= K) v = 0.0f;
      As[m][k] = f2bf(v);
    }
    if (transB) {
#pragma unroll 1
      for (int i = 0; i < (BN * BK) / 256; ++i) {
        int idx = tid + i * 256;
        int n = idx >> 5, k = idx & 31;
        int gn = blockN + n, gk = k0 + k;
        int cn = gn < N ? gn : N - 1;
        int ck = gk < K ? gk : K - 1;
        float v = B[(size_t)cn * ldb + ck];
        if (gn >= N || gk >= K) v = 0.0f;
        Bs[n][k] = f2bf(v);
      }
    } else {
#pragma unroll 1
      for (int i = 0; i < (BN * BK) / 256; ++i) {
        int idx = tid + i * 256;
        int k = idx >> 7, n = idx & 127;
        int gn = blockN + n, gk = k0 + k;
        int cn = gn < N ? gn : N - 1;
        int ck = gk < K ? gk : K - 1;
        float v = B[(size_t)ck * ldb + cn];
        if (gn >= N || gk >= K) v = 0.0f;
        Bs[n][k] = f2bf(v);
      }
    }
  };

  // Fast unguarded vectorized tile load (hot path).
  auto load_fast = [&](int k0) {
#pragma unroll
    for (int i = 0; i < 4; ++i) {
      int idx = tid + i * 256;
      int m = idx >> 3, kq = (idx & 7) * 4;
      const float4 v = *(const float4*)(A + (size_t)(blockM + m) * lda + k0 + kq);
      uint2 pp;
      pp.x = pack2bf(v.x, v.y);
      pp.y = pack2bf(v.z, v.w);
      *(uint2*)&As[m][kq] = pp;
    }
    if (transB) {
#pragma unroll
      for (int i = 0; i < 4; ++i) {
        int idx = tid + i * 256;
        int n = idx >> 3, kq = (idx & 7) * 4;
        const float4 v = *(const float4*)(B + (size_t)(blockN + n) * ldb + k0 + kq);
        uint2 pp;
        pp.x = pack2bf(v.x, v.y);
        pp.y = pack2bf(v.z, v.w);
        *(uint2*)&Bs[n][kq] = pp;
      }
    } else {
#pragma unroll
      for (int i = 0; i < 4; ++i) {
        int idx = tid + i * 256;
        int k = idx >> 5, nq = (idx & 31) * 4;
        const float4 v = *(const float4*)(B + (size_t)(k0 + k) * ldb + blockN + nq);
        Bs[nq + 0][k] = f2bf(v.x);
        Bs[nq + 1][k] = f2bf(v.y);
        Bs[nq + 2][k] = f2bf(v.z);
        Bs[nq + 3][k] = f2bf(v.w);
      }
    }
  };

  if (interior) {
#pragma unroll 1
    for (int ks = 0; ks < kfull; ++ks) {
      const int k0 = ks * BK;
      load_fast(k0);
      // prefetch next k-step's tile rows while WMMAs run
      if (ks + 1 < kfull) {
        const int kn = k0 + BK;
        if (tid < 128) {
          __builtin_prefetch(A + (size_t)(blockM + tid) * lda + kn, 0, 3);
        } else {
          int t = tid - 128;
          if (transB)
            __builtin_prefetch(B + (size_t)(blockN + t) * ldb + kn, 0, 3);
          else if (t < BK)
            __builtin_prefetch(B + (size_t)(kn + t) * ldb + blockN, 0, 3);
        }
      }
      __syncthreads();
      mma_step(As, Bs, acc, waveM, waveN, l16, half);
      __syncthreads();
    }
    if (ksteps > kfull) {   // K tail (zero-padded)
      load_guarded(kfull * BK);
      __syncthreads();
      mma_step(As, Bs, acc, waveM, waveN, l16, half);
      __syncthreads();
    }
  } else {
#pragma unroll 1
    for (int ks = 0; ks < ksteps; ++ks) {
      load_guarded(ks * BK);
      __syncthreads();
      mma_step(As, Bs, acc, waveM, waveN, l16, half);
      __syncthreads();
    }
  }

  // ---- epilogue: C/D layout -> row = p + 8*half, col = l16 ----
  if (interior) {
#pragma unroll
    for (int tm = 0; tm < 4; ++tm)
#pragma unroll
      for (int tn = 0; tn < 2; ++tn)
#pragma unroll
        for (int p = 0; p < 8; ++p) {
          int row = blockM + waveM * 64 + tm * 16 + half * 8 + p;
          int col = blockN + waveN * 32 + tn * 16 + l16;
          float v = acc[tm][tn][p];
          if (bias) v += bias[col];
          if (doRelu) v = v > 0.0f ? v : 0.0f;
          C[(size_t)row * ldc + col] = v;
        }
  } else {
#pragma unroll
    for (int tm = 0; tm < 4; ++tm)
#pragma unroll
      for (int tn = 0; tn < 2; ++tn)
#pragma unroll
        for (int p = 0; p < 8; ++p) {
          int row = blockM + waveM * 64 + tm * 16 + half * 8 + p;
          int col = blockN + waveN * 32 + tn * 16 + l16;
          if (row < M && col < N) {
            float v = acc[tm][tn][p];
            if (bias) v += bias[col];
            if (doRelu) v = v > 0.0f ? v : 0.0f;
            C[(size_t)row * ldc + col] = v;
          }
        }
  }
}

// ---------------------------------------------------------------------------
// Transpose: out[c*rows + r] = in[r*cols + c]
// ---------------------------------------------------------------------------
__global__ void transpose_kernel(const float* __restrict__ in,
                                 float* __restrict__ out, int rows, int cols) {
  __shared__ float tile[32][33];
  int bx = blockIdx.x * 32, by = blockIdx.y * 32;
  int x = bx + threadIdx.x, y = by + threadIdx.y;
#pragma unroll
  for (int j = 0; j < 32; j += 8)
    if (x < cols && (y + j) < rows)
      tile[threadIdx.y + j][threadIdx.x] = in[(size_t)(y + j) * cols + x];
  __syncthreads();
  x = by + threadIdx.x;  // output col = input row
  y = bx + threadIdx.y;  // output row = input col
#pragma unroll
  for (int j = 0; j < 32; j += 8)
    if (x < rows && (y + j) < cols)
      out[(size_t)(y + j) * rows + x] = tile[threadIdx.x][threadIdx.y + j];
}

// ---------------------------------------------------------------------------
// Scores: s[i] = Z[i,:].vt + vtb ; r[i] = Z[i,:].vr + vrb  (one block per row)
// ---------------------------------------------------------------------------
__global__ void score_kernel(const float* __restrict__ Z, int f,
                             const float* __restrict__ vt, const float* __restrict__ vtb,
                             const float* __restrict__ vr, const float* __restrict__ vrb,
                             float* __restrict__ s, float* __restrict__ r) {
  __shared__ float reds[256], redr[256];
  int row = blockIdx.x;
  const float* zr = Z + (size_t)row * f;
  float as = 0.f, ar = 0.f;
  for (int c = threadIdx.x; c < f; c += 256) {
    float z = zr[c];
    as += z * vt[c];
    ar += z * vr[c];
  }
  reds[threadIdx.x] = as; redr[threadIdx.x] = ar;
  __syncthreads();
  for (int off = 128; off > 0; off >>= 1) {
    if (threadIdx.x < off) {
      reds[threadIdx.x] += reds[threadIdx.x + off];
      redr[threadIdx.x] += redr[threadIdx.x + off];
    }
    __syncthreads();
  }
  if (threadIdx.x == 0) { s[row] = reds[0] + vtb[0]; r[row] = redr[0] + vrb[0]; }
}

// ---------------------------------------------------------------------------
// Att[i,j] = softmax_j( sigmoid(A[i,j]*s[j] + A[j,i]*r[i]) )
// sigmoid output is in (0,1) -> exp never overflows -> single-pass softmax,
// row held in registers (16 regs * 256 threads covers n<=4096).
// ---------------------------------------------------------------------------
__global__ __launch_bounds__(256) void att_kernel(
    const float* __restrict__ A, const float* __restrict__ AT,
    const float* __restrict__ s, const float* __restrict__ r,
    float* __restrict__ Att, int n) {
  __shared__ float red[256];
  int i = blockIdx.x;
  float ri = r[i];
  const float* Ar  = A  + (size_t)i * n;
  const float* ATr = AT + (size_t)i * n;
  float vals[16];
  float part = 0.f;
#pragma unroll
  for (int e = 0; e < 16; ++e) {
    int j = threadIdx.x + e * 256;
    float v = 0.f;
    if (j < n) {
      float x = Ar[j] * s[j] + ATr[j] * ri;
      float sig = 1.0f / (1.0f + __expf(-x));
      v = __expf(sig);
    }
    vals[e] = v;
    part += v;
  }
  red[threadIdx.x] = part;
  __syncthreads();
  for (int off = 128; off > 0; off >>= 1) {
    if (threadIdx.x < off) red[threadIdx.x] += red[threadIdx.x + off];
    __syncthreads();
  }
  float inv = 1.0f / red[0];
  float* Or = Att + (size_t)i * n;
#pragma unroll
  for (int e = 0; e < 16; ++e) {
    int j = threadIdx.x + e * 256;
    if (j < n) Or[j] = vals[e] * inv;
  }
}

// ---------------------------------------------------------------------------
// D (M x 64) = concat([gG (64 x G), H3.T (64 x N)], axis=1).T
//   D[m,c] = gG[c, m]          for m < G
//   D[m,c] = H3[m - G, c]      for m >= G   (H3 is N x 64)
// ---------------------------------------------------------------------------
__global__ void build_d_kernel(const float* __restrict__ gG,
                               const float* __restrict__ H3,
                               float* __restrict__ D) {
  int idx = blockIdx.x * blockDim.x + threadIdx.x;
  if (idx >= MC * 64) return;
  int m = idx >> 6, c = idx & 63;
  D[idx] = (m < GC) ? gG[(size_t)c * GC + m] : H3[(size_t)(m - GC) * 64 + c];
}

// ---------------------------------------------------------------------------
// Host-side pipeline
// ---------------------------------------------------------------------------
static void launch_gemm(hipStream_t st, const float* A, const float* B, float* C,
                        const float* bias, int M, int N, int K,
                        int lda, int ldb, int ldc, bool transB, bool relu) {
  dim3 g((N + BN - 1) / BN, (M + BM - 1) / BM);
  gemm_bf16_wmma<<<g, 256, 0, st>>>(A, B, C, bias, M, N, K, lda, ldb, ldc,
                                    transB ? 1 : 0, relu ? 1 : 0);
}

extern "C" void kernel_launch(void* const* d_in, const int* in_sizes, int n_in,
                              void* d_out, int out_size, void* d_ws, size_t ws_size,
                              hipStream_t stream) {
  (void)in_sizes; (void)n_in; (void)out_size; (void)ws_size;

  const float* X     = (const float*)d_in[0];
  const float* A_enc = (const float*)d_in[1];
  const float* A_dec = (const float*)d_in[2];
  const float* gG    = (const float*)d_in[3];
  // layer order: e1,e2,e3,d1,d2,d3,d4 ; comp: 0=W,1=Wb,2=vt,3=vtb,4=vr,5=vrb
  auto L = [&](int layer, int comp) { return (const float*)d_in[4 + layer * 6 + comp]; };

  float* ws   = (float*)d_ws;
  float* ws0  = ws;                   // 16.0M floats : X^T (encoder) / A_dec^T (decoder)
  float* ws1  = ws0 + 16000000;       //  9.0M floats : A_enc^T
  float* ws2  = ws1 + 9000000;        // 16.0M floats : Att
  float* ws3  = ws2 + 16000000;       // 16.0M floats : Z
  float* ws4  = ws3 + 16000000;       //  2.048M      : H ping
  float* ws5  = ws4 + 2048000;        //  2.048M      : H pong
  float* sbuf = ws5 + 2048000;        //  4096
  float* rbuf = sbuf + 4096;          //  4096   (total ~244 MB)

  float* out_cell = (float*)d_out;               // (M x M)
  float* out_gene = out_cell + (size_t)MC * MC;  // (M x G)

  // One GAT layer: Z = relu(Hin*W^T + b); s,r; Att; Hout = Att*Z
  auto gat = [&](const float* A, const float* AT, int n,
                 const float* Hin, int fin, int layer, int fout,
                 float* Hout) {
    launch_gemm(stream, Hin, L(layer, 0), ws3, L(layer, 1),
                n, fout, fin, fin, fin, fout, /*transB=*/true, /*relu=*/true);
    score_kernel<<<n, 256, 0, stream>>>(ws3, fout, L(layer, 2), L(layer, 3),
                                        L(layer, 4), L(layer, 5), sbuf, rbuf);
    att_kernel<<<n, 256, 0, stream>>>(A, AT, sbuf, rbuf, ws2, n);
    launch_gemm(stream, ws2, ws3, Hout, nullptr,
                n, fout, n, n, fout, fout, /*transB=*/false, /*relu=*/false);
  };

  dim3 tb(32, 8);
  // X^T and A_enc^T
  transpose_kernel<<<dim3((NC + 31) / 32, (NC + 31) / 32), tb, 0, stream>>>(X, ws0, NC, NC);
  transpose_kernel<<<dim3((NC + 31) / 32, (NC + 31) / 32), tb, 0, stream>>>(A_enc, ws1, NC, NC);

  // encoder
  gat(A_enc, ws1, NC, ws0, NC,  0, 512, ws4);   // H1 = ws4 (3000x512)
  gat(A_enc, ws1, NC, ws4, 512, 1, 256, ws5);   // H2 = ws5 (3000x256)
  gat(A_enc, ws1, NC, ws5, 256, 2,  64, ws4);   // H3 = ws4 (3000x64)

  // decoder input D = ws5 (4000x64); A_dec^T overwrites X^T slot
  build_d_kernel<<<(MC * 64 + 255) / 256, 256, 0, stream>>>(gG, ws4, ws5);
  transpose_kernel<<<dim3((MC + 31) / 32, (MC + 31) / 32), tb, 0, stream>>>(A_dec, ws0, MC, MC);

  // decoder
  gat(A_dec, ws0, MC, ws5, 64,  3, 256, ws4);       // D1 = ws4 (4000x256)
  gat(A_dec, ws0, MC, ws4, 256, 4, 512, ws5);       // D2 = ws5 (4000x512)
  gat(A_dec, ws0, MC, ws5, 512, 5, MC,  out_cell);  // cell (4000x4000)
  gat(A_dec, ws0, MC, ws5, 512, 6, GC,  out_gene);  // gene (4000x1000)
}
